// DiagonalLinear_463856468196
// MI455X (gfx1250) — matrix operand order repaired and checked by reading, not compiled, not run on hardware
//
#include <hip/hip_runtime.h>

#define N_COPIES 64
#define IP_DIM   256
#define OP_DIM   256
#define BATCH    4096

#define M_BLK 128
#define N_BLK 128
#define K_BLK 64
#define LDS_STRIDE 72   // bf16 elems; 144B row stride: 16B aligned, rotates banks by 4 dwords/row

#define WP_BYTES ((size_t)N_COPIES * OP_DIM * IP_DIM * 2)   // 8,388,608 B prepacked bf16 W

typedef __attribute__((ext_vector_type(16))) __bf16 v16bf;
typedef __attribute__((ext_vector_type(8)))  float  v8f;

union Frag {
  v16bf v;
  uint4 q[2];
};

// fp32 -> bf16 (RNE), packed pair -> one dword.
// v_bfe + v_add3 per float, then one v_perm_b32 to merge the two high halves.
__device__ __forceinline__ unsigned pack2_bf16(float lo, float hi) {
  unsigned a = __float_as_uint(lo);
  unsigned b = __float_as_uint(hi);
  a += 0x7FFFu + ((a >> 16) & 1u);
  b += 0x7FFFu + ((b >> 16) & 1u);
  // result bytes: [0]=a.b2 [1]=a.b3 [2]=b.b2 [3]=b.b3  (src0=b -> codes 4..7, src1=a -> 0..3)
  return __builtin_amdgcn_perm(b, a, 0x07060302u);
}

// ---------------------------------------------------------------------------
// W prepack: fp32 [copy][o][k] -> bf16 B-fragments.
// Fragment f = (copy*16 + o/16)*8 + k/32 ; 1024 B each.
// Lane L of the consuming wave reads 32 B at f*1024 + L*32:
//   o = (o/16)*16 + (L&15), K = 16 contiguous values starting at (k/32)*32 + (L>>4)*16
// ---------------------------------------------------------------------------
__global__ __launch_bounds__(256)
void prepack_W_frag(const float* __restrict__ W, uint4* __restrict__ Wp) {
  int gid = blockIdx.x * 256 + threadIdx.x;   // one thread per (fragment, lane)
  int L   = gid & 31;
  int f   = gid >> 5;
  int ks  = f & 7;
  int j16 = (f >> 3) & 15;
  int cp  = f >> 7;
  int o   = j16 * 16 + (L & 15);
  int k0  = ks * 32 + ((L >> 4) << 4);

  const float* src = W + ((size_t)cp * OP_DIM + o) * IP_DIM + k0;  // 16 floats
  unsigned d[8];
  #pragma unroll
  for (int t = 0; t < 4; ++t) {
    float4 v = *(const float4*)(src + 4 * t);
    d[2 * t]     = pack2_bf16(v.x, v.y);
    d[2 * t + 1] = pack2_bf16(v.z, v.w);
  }
  uint4* dst = Wp + (size_t)f * 64 + L * 2;
  dst[0] = make_uint4(d[0], d[1], d[2], d[3]);
  dst[1] = make_uint4(d[4], d[5], d[6], d[7]);
}

// ---------------------------------------------------------------------------
// Main GEMM, prepacked-W path: x staged fp32->bf16 through LDS, B fragments
// loaded directly from L2-resident prepacked W (2x global_load_b128 / frag).
// ---------------------------------------------------------------------------
__global__ __launch_bounds__(256, 2)
void block_diag_gemm_wp(const float* __restrict__ x,
                        const uint4* __restrict__ Wp,
                        const float* __restrict__ bias,
                        float* __restrict__ out) {
  __shared__ __align__(16) unsigned short Xs[M_BLK][LDS_STRIDE];

  const int tid  = threadIdx.x;
  const int lane = tid & 31;
  const int wave = tid >> 5;
  const int wm   = wave & 3;      // 0..3 -> 32-row slab of M tile
  const int wn   = wave >> 2;     // 0..1 -> 64-col slab of N tile
  const int halo = lane >> 4;
  const int l15  = lane & 15;

  const int m0   = blockIdx.x * M_BLK;
  const int n0   = blockIdx.y * N_BLK;
  const int copy = blockIdx.z;

  const size_t ROWLEN = (size_t)N_COPIES * IP_DIM;   // 16384
  const float* xp = x + (size_t)copy * IP_DIM;

  v8f acc[2][4];
  #pragma unroll
  for (int i = 0; i < 2; ++i)
    #pragma unroll
    for (int j = 0; j < 4; ++j) {
      v8f z = {};
      acc[i][j] = z;
    }

  for (int kk = 0; kk < IP_DIM; kk += K_BLK) {
    // ---- stage X tile (128 x 64 fp32 -> bf16) ----
    #pragma unroll
    for (int j = 0; j < 8; ++j) {
      int id   = tid + 256 * j;          // 0..2047
      int row  = id >> 4;                // 0..127
      int col4 = (id & 15) << 2;         // 0,4,...,60
      float4 xv = *(const float4*)(xp + (size_t)(m0 + row) * ROWLEN + kk + col4);
      unsigned* xd = (unsigned*)&Xs[row][col4];
      xd[0] = pack2_bf16(xv.x, xv.y);
      xd[1] = pack2_bf16(xv.z, xv.w);
    }
    __syncthreads();

    #pragma unroll
    for (int ki = 0; ki < K_BLK; ki += 32) {
      const int ks = (kk + ki) >> 5;     // global K-step index 0..7
      Frag a[2], b[4];
      #pragma unroll
      for (int i = 0; i < 2; ++i) {
        const unsigned short* xr = &Xs[wm * 32 + i * 16 + l15][0];
        int kb = ki + (halo << 4);
        a[i].q[0] = *(const uint4*)(xr + kb);
        a[i].q[1] = *(const uint4*)(xr + kb + 8);
      }
      #pragma unroll
      for (int j = 0; j < 4; ++j) {
        int jg = (n0 >> 4) + wn * 4 + j;                       // global 16-col subtile
        size_t fi = (((size_t)copy * 16 + jg) * 8 + ks) * 64 + lane * 2;
        b[j].q[0] = Wp[fi];
        b[j].q[1] = Wp[fi + 1];
      }
      #pragma unroll
      for (int i = 0; i < 2; ++i)
        #pragma unroll
        for (int j = 0; j < 4; ++j)
          acc[i][j] = __builtin_amdgcn_wmma_f32_16x16x32_bf16(
              false, a[i].v, false, b[j].v, (short)0, acc[i][j], false, false);
    }
    __syncthreads();
  }

  #pragma unroll
  for (int j = 0; j < 4; ++j) {
    int ocol = n0 + wn * 64 + j * 16 + l15;
    float bj = bias[copy * OP_DIM + ocol];
    #pragma unroll
    for (int i = 0; i < 2; ++i) {
      int rbase = m0 + wm * 32 + i * 16 + (halo << 3);
      #pragma unroll
      for (int r = 0; r < 8; ++r) {
        float* p = out + (size_t)(rbase + r) * ROWLEN + copy * OP_DIM + ocol;
        __builtin_nontemporal_store(acc[i][j][r] + bj, p);
      }
    }
  }
}

// ---------------------------------------------------------------------------
// Fallback: fused conversion of both x and W (used when d_ws is too small).
// ---------------------------------------------------------------------------
__global__ __launch_bounds__(256, 2)
void block_diag_linear_fused(const float* __restrict__ x,
                             const float* __restrict__ W,
                             const float* __restrict__ bias,
                             float* __restrict__ out) {
  __shared__ __align__(16) unsigned short Xs[M_BLK][LDS_STRIDE];
  __shared__ __align__(16) unsigned short Ws[N_BLK][LDS_STRIDE];

  const int tid  = threadIdx.x;
  const int lane = tid & 31;
  const int wave = tid >> 5;
  const int wm   = wave & 3;
  const int wn   = wave >> 2;
  const int halo = lane >> 4;
  const int l15  = lane & 15;

  const int m0   = blockIdx.x * M_BLK;
  const int n0   = blockIdx.y * N_BLK;
  const int copy = blockIdx.z;

  const size_t ROWLEN = (size_t)N_COPIES * IP_DIM;
  const float* xp = x + (size_t)copy * IP_DIM;
  const float* wp = W + (size_t)copy * OP_DIM * IP_DIM;

  v8f acc[2][4];
  #pragma unroll
  for (int i = 0; i < 2; ++i)
    #pragma unroll
    for (int j = 0; j < 4; ++j) {
      v8f z = {};
      acc[i][j] = z;
    }

  for (int kk = 0; kk < IP_DIM; kk += K_BLK) {
    #pragma unroll
    for (int j = 0; j < 8; ++j) {
      int id   = tid + 256 * j;
      int row  = id >> 4;
      int col4 = (id & 15) << 2;
      float4 xv = *(const float4*)(xp + (size_t)(m0 + row) * ROWLEN + kk + col4);
      float4 wv = *(const float4*)(wp + (size_t)(n0 + row) * IP_DIM + kk + col4);
      unsigned* xd = (unsigned*)&Xs[row][col4];
      xd[0] = pack2_bf16(xv.x, xv.y);
      xd[1] = pack2_bf16(xv.z, xv.w);
      unsigned* wd = (unsigned*)&Ws[row][col4];
      wd[0] = pack2_bf16(wv.x, wv.y);
      wd[1] = pack2_bf16(wv.z, wv.w);
    }
    __syncthreads();

    #pragma unroll
    for (int ki = 0; ki < K_BLK; ki += 32) {
      Frag a[2], b[4];
      #pragma unroll
      for (int i = 0; i < 2; ++i) {
        const unsigned short* xr = &Xs[wm * 32 + i * 16 + l15][0];
        int kb = ki + (halo << 4);
        a[i].q[0] = *(const uint4*)(xr + kb);
        a[i].q[1] = *(const uint4*)(xr + kb + 8);
      }
      #pragma unroll
      for (int j = 0; j < 4; ++j) {
        const unsigned short* wr = &Ws[wn * 64 + j * 16 + l15][0];
        int kb = ki + (halo << 4);
        b[j].q[0] = *(const uint4*)(wr + kb);
        b[j].q[1] = *(const uint4*)(wr + kb + 8);
      }
      #pragma unroll
      for (int i = 0; i < 2; ++i)
        #pragma unroll
        for (int j = 0; j < 4; ++j)
          acc[i][j] = __builtin_amdgcn_wmma_f32_16x16x32_bf16(
              false, a[i].v, false, b[j].v, (short)0, acc[i][j], false, false);
    }
    __syncthreads();
  }

  #pragma unroll
  for (int j = 0; j < 4; ++j) {
    int ocol = n0 + wn * 64 + j * 16 + l15;
    float bj = bias[copy * OP_DIM + ocol];
    #pragma unroll
    for (int i = 0; i < 2; ++i) {
      int rbase = m0 + wm * 32 + i * 16 + (halo << 3);
      #pragma unroll
      for (int r = 0; r < 8; ++r) {
        float* p = out + (size_t)(rbase + r) * ROWLEN + copy * OP_DIM + ocol;
        __builtin_nontemporal_store(acc[i][j][r] + bj, p);
      }
    }
  }
}

extern "C" void kernel_launch(void* const* d_in, const int* in_sizes, int n_in,
                              void* d_out, int out_size, void* d_ws, size_t ws_size,
                              hipStream_t stream) {
  (void)in_sizes; (void)n_in; (void)out_size;
  const float* x  = (const float*)d_in[0];
  const float* W  = (const float*)d_in[1];
  const float* b  = (const float*)d_in[2];
  float* out = (float*)d_out;

  dim3 grid(BATCH / M_BLK, OP_DIM / N_BLK, N_COPIES);  // (32, 2, 64)
  dim3 block(256);

  if (ws_size >= WP_BYTES) {
    uint4* Wp = (uint4*)d_ws;
    int nfrag_threads = N_COPIES * 16 * 8 * 32;        // 262144
    prepack_W_frag<<<nfrag_threads / 256, 256, 0, stream>>>(W, Wp);
    block_diag_gemm_wp<<<grid, block, 0, stream>>>(x, Wp, b, out);
  } else {
    block_diag_linear_fused<<<grid, block, 0, stream>>>(x, W, b, out);
  }
}
